// ScaledDotproductAttention_47820165874253
// MI455X (gfx1250) — matrix operand order repaired and checked by reading, not compiled, not run on hardware
//
#include <hip/hip_runtime.h>

typedef __attribute__((ext_vector_type(16))) __bf16 v16bf;
typedef __attribute__((ext_vector_type(8)))  float  v8f;

#define B_DIM   16
#define S_DIM   2048
#define D_DIM   128
#define TQ      16                 // q rows per workgroup
#define WAVES   4
#define THREADS (WAVES * 32)
#define NTILES  (S_DIM / 16)       // 128 WMMA n-tiles across k dimension

__global__ __launch_bounds__(THREADS)
void sdpa_wmma_kernel(const float* __restrict__ Q,
                      const float* __restrict__ K,
                      const float* __restrict__ V,
                      float* __restrict__ Out,   // [B,S,D]
                      float* __restrict__ Attn)  // [B,S,S]
{
    // 16 x 2048 score/probability tile, f32 (128 KB of the 320 KB WGP LDS)
    __shared__ float sP[TQ][S_DIM];
    __shared__ float sRed[TQ][8];
    __shared__ float sM[TQ];
    __shared__ float sR[TQ];

    const int tid  = threadIdx.x;
    const int lane = tid & 31;
    const int wave = tid >> 5;
    const int b    = blockIdx.x / (S_DIM / TQ);
    const int q0   = (blockIdx.x % (S_DIM / TQ)) * TQ;

    const float* Qb = Q + (size_t)b * S_DIM * D_DIM;
    const float* Kb = K + (size_t)b * S_DIM * D_DIM;
    const float* Vb = V + (size_t)b * S_DIM * D_DIM;

    const int half = lane >> 4;   // 0: lanes 0-15, 1: lanes 16-31
    const int l16  = lane & 15;

    // ---------------- Preload Q A-fragments (16x32 bf16 per K-chunk) --------
    // A layout: lanes 0-15 -> M=lane, K in {0..7, 16..23}; lanes 16-31 -> M=lane-16,
    // K in {8..15, 24..31}.  half*8 base gives exactly that split.
    v16bf aQ[4];
    {
        const float* qrow = Qb + (size_t)(q0 + l16) * D_DIM;
        #pragma unroll
        for (int c = 0; c < 4; ++c) {
            const float* qr = qrow + c * 32 + half * 8;
            #pragma unroll
            for (int i = 0; i < 8; ++i) {
                aQ[c][i]     = (__bf16)qr[i];
                aQ[c][i + 8] = (__bf16)qr[i + 16];
            }
        }
    }

    // ---------------- GEMM1: scores = Q * K^T, masked + scaled into LDS -----
    const float scale = 0.08838834764831845f;  // 1/sqrt(128)
    for (int nt = wave; nt < NTILES; nt += WAVES) {
        const int n0 = nt * 16;
        // speculative prefetch of this wave's next K tile (dropped if OOB)
        if (nt + WAVES < NTILES) {
            __builtin_prefetch(Kb + (size_t)(n0 + WAVES * 16 + l16) * D_DIM, 0, 3);
        }
        v8f c = {};
        #pragma unroll
        for (int kc = 0; kc < 4; ++kc) {
            // B layout (32x16): lane -> column N = l16, K = kc*32 + half*8 + {0..7, 16..23}.
            // B[k][n] = K_mat[n0+n][k]  -> contiguous along D.
            const float* kr = Kb + (size_t)(n0 + l16) * D_DIM + kc * 32 + half * 8;
            v16bf bK;
            #pragma unroll
            for (int i = 0; i < 16; ++i) bK[i] = (__bf16)kr[(i & 7) + ((i >> 3) << 4)];
            c = __builtin_amdgcn_wmma_f32_16x16x32_bf16(
                    false, aQ[kc], false, bK, (short)0, c, false, false);
        }
        // C/D layout: lane L, vgpr v -> (M = v + 8*(L>=16), N = L%16)
        #pragma unroll
        for (int v = 0; v < 8; ++v) {
            const int M  = v + half * 8;
            const int kg = n0 + l16;
            // keep strictly-upper (k > q); otherwise write 0 (reference masks with 0)
            sP[M][kg] = (kg > q0 + M) ? c[v] * scale : 0.0f;
        }
    }
    __syncthreads();

    // ---------------- Softmax over k (rows of the 16x2048 tile) ------------
    const int row   = tid & 15;
    const int chunk = tid >> 4;         // 8 chunks x 256 cols per row
    {   // per-row max (vectorized LDS scan)
        const float4* rowv = (const float4*)&sP[row][chunk * 256];
        float m = -3.402823466e38f;
        for (int i = 0; i < 64; ++i) {
            float4 x = rowv[i];
            m = fmaxf(m, fmaxf(fmaxf(x.x, x.y), fmaxf(x.z, x.w)));
        }
        sRed[row][chunk] = m;
    }
    __syncthreads();
    if (tid < TQ) {
        float m = sRed[tid][0];
        #pragma unroll
        for (int i = 1; i < 8; ++i) m = fmaxf(m, sRed[tid][i]);
        sM[tid] = m;
    }
    __syncthreads();
    {   // exp(x - m) stored back into LDS; accumulate partial row sum
        const float m = sM[row];
        float4* rowv = (float4*)&sP[row][chunk * 256];
        float s = 0.0f;
        for (int i = 0; i < 64; ++i) {
            float4 x = rowv[i];
            x.x = __expf(x.x - m); x.y = __expf(x.y - m);
            x.z = __expf(x.z - m); x.w = __expf(x.w - m);
            s += (x.x + x.y) + (x.z + x.w);
            rowv[i] = x;
        }
        sRed[row][chunk] = s;
    }
    __syncthreads();
    if (tid < TQ) {
        float s = 0.0f;
        #pragma unroll
        for (int i = 0; i < 8; ++i) s += sRed[tid][i];
        sR[tid] = 1.0f / s;
    }
    __syncthreads();

    // ------- Normalize + store attention: b128 stores on the 268MB stream ---
    {
        float* attnBase = Attn + ((size_t)b * S_DIM + q0) * S_DIM;
        for (int idx = tid; idx < TQ * (S_DIM / 4); idx += THREADS) {
            const int r  = idx >> 9;            // S_DIM/4 == 512
            const int c4 = idx & 511;
            float4 e = ((float4*)sP[r])[c4];
            const float rs = sR[r];
            e.x *= rs; e.y *= rs; e.z *= rs; e.w *= rs;
            ((float4*)sP[r])[c4] = e;                              // probs for GEMM2
            ((float4*)(attnBase + (size_t)r * S_DIM))[c4] = e;     // global_store_b128
        }
    }
    __syncthreads();

    // ---------------- GEMM2: O = P * V  (16x2048 @ 2048x128) ---------------
    const int d0 = wave * 32;                    // each wave: 2 d-tiles of 16
    v8f o0 = {}, o1 = {};
    #pragma unroll 2
    for (int kk = 0; kk < S_DIM; kk += 32) {
        // A fragment from LDS probabilities
        v16bf aP;
        const float* pr = &sP[l16][kk + half * 8];
        #pragma unroll
        for (int i = 0; i < 8; ++i) {
            aP[i]     = (__bf16)pr[i];
            aP[i + 8] = (__bf16)pr[i + 16];
        }
        // B fragments from V: lane -> column d0+l16, K rows kk+half*8 + {0..7, 16..23}
        const float* vc = Vb + (size_t)(kk + half * 8) * D_DIM + d0 + l16;
        v16bf b0, b1;
        #pragma unroll
        for (int i = 0; i < 16; ++i) {
            b0[i] = (__bf16)vc[(size_t)((i & 7) + ((i >> 3) << 4)) * D_DIM];
            b1[i] = (__bf16)vc[(size_t)((i & 7) + ((i >> 3) << 4)) * D_DIM + 16];
        }
        o0 = __builtin_amdgcn_wmma_f32_16x16x32_bf16(
                 false, aP, false, b0, (short)0, o0, false, false);
        o1 = __builtin_amdgcn_wmma_f32_16x16x32_bf16(
                 false, aP, false, b1, (short)0, o1, false, false);
    }
    {
        float* outBase = Out + ((size_t)b * S_DIM + q0) * D_DIM;
        #pragma unroll
        for (int v = 0; v < 8; ++v) {
            const int M = v + half * 8;
            outBase[(size_t)M * D_DIM + d0 + l16]      = o0[v];
            outBase[(size_t)M * D_DIM + d0 + 16 + l16] = o1[v];
        }
    }
}

extern "C" void kernel_launch(void* const* d_in, const int* in_sizes, int n_in,
                              void* d_out, int out_size, void* d_ws, size_t ws_size,
                              hipStream_t stream) {
    const float* Wq = (const float*)d_in[0];
    const float* Wk = (const float*)d_in[1];
    const float* Wv = (const float*)d_in[2];
    float* out  = (float*)d_out;                                   // [B,S,D]
    float* attn = out + (size_t)B_DIM * S_DIM * D_DIM;             // [B,S,S]

    dim3 grid(B_DIM * (S_DIM / TQ));   // 16 * 128 = 2048 workgroups
    dim3 block(THREADS);               // 4 waves of 32
    sdpa_wmma_kernel<<<grid, block, 0, stream>>>(Wq, Wk, Wv, out, attn);
}
